// SELAREncoder_55920474194406
// MI455X (gfx1250) — matrix operand order repaired and checked by reading, not compiled, not run on hardware
//
#include <hip/hip_runtime.h>
#include <hip/hip_bf16.h>

typedef __attribute__((ext_vector_type(16))) _Float16 v16h;
typedef __attribute__((ext_vector_type(8)))  _Float16 v8h;
typedef __attribute__((ext_vector_type(4)))  _Float16 v4h;
typedef __attribute__((ext_vector_type(8)))  float    v8f;

#define HIDF 128   // feature width (K dim of every GEMM)

// ---------------------------------------------------------------------------
// Pack fp32 W[128 x NOUT] into f16 WMMA B-fragments:
// frag[((tile*4 + ks)*32 + lane)*16 + i] = W[(ks*32 + (lane>>4)*8 + i + (i>=8)*8)*NOUT
//                                            + tile*16 + (lane&15)]
// ---------------------------------------------------------------------------
__global__ void pack_w_k(const float* __restrict__ W, _Float16* __restrict__ out,
                         int NOUT) {
  int tid   = blockIdx.x * blockDim.x + threadIdx.x;
  int total = (NOUT / 16) * 4 * 32;
  if (tid >= total) return;
  int lane = tid & 31;
  int ks   = (tid >> 5) & 3;
  int tile = tid >> 7;
  int k0 = ks * 32 + (lane >> 4) * 8;
  int n  = tile * 16 + (lane & 15);
  _Float16* o = out + (size_t)tid * 16;
#pragma unroll
  for (int i = 0; i < 16; ++i) o[i] = (_Float16)W[(size_t)(k0 + i + (i >= 8 ? 8 : 0)) * NOUT + n];
}

// fp32 -> f16 row-major copy
__global__ void cvt16_k(const float* __restrict__ X, _Float16* __restrict__ Xh, int n32) {
  int t = blockIdx.x * blockDim.x + threadIdx.x;
  if (t >= n32) return;
  float4 p = *(const float4*)(X + ((size_t)t << 2));
  v4h o = { (_Float16)p.x, (_Float16)p.y, (_Float16)p.z, (_Float16)p.w };
  *(v4h*)(Xh + ((size_t)t << 2)) = o;
}

// ---------------------------------------------------------------------------
// Shared WMMA body: 16-row M tile x 16-col N tile, K=128 in 4 WMMA steps.
// A: two aligned v8h loads/step (f16 row-major); B: one aligned 32B load/step
// from the packed-fragment buffer.
// ---------------------------------------------------------------------------
__device__ __forceinline__ v8f wmma_tile(const _Float16* __restrict__ xr,
                                         const _Float16* __restrict__ wf) {
  v8f c = {};
#pragma unroll
  for (int ks = 0; ks < HIDF / 32; ++ks) {
    v8h alo = *(const v8h*)(xr + ks * 32);
    v8h ahi = *(const v8h*)(xr + ks * 32 + 16);
    v16h a = __builtin_shufflevector(alo, ahi, 0, 1, 2, 3, 4, 5, 6, 7,
                                     8, 9, 10, 11, 12, 13, 14, 15);
    v16h b = *(const v16h*)(wf + (size_t)ks * 32 * 16);
    c = __builtin_amdgcn_wmma_f32_16x16x32_f16(false, a, false, b,
                                               (short)0, c, false, false);
  }
  return c;
}

// ---------------------------------------------------------------------------
// GCN GEMM with fused self-loop init:
//   P = Xh @ W            (raw product, consumed by edge gather)
//   G = P * dinv^2 + b    (aggregation buffer seeded with self-loop + bias)
// 8 waves per block, NOUT = 128.
// ---------------------------------------------------------------------------
__global__ void gemm_gcn_k(const _Float16* __restrict__ Xh,
                           const _Float16* __restrict__ Wp,
                           const float* __restrict__ bias,
                           const float* __restrict__ dinv,
                           float* __restrict__ P, float* __restrict__ G,
                           int nrows) {
  const int wave   = threadIdx.x >> 5;
  const int lane   = threadIdx.x & 31;
  const int mbase  = blockIdx.x * 16;
  const int ln     = lane & 15;
  const int hiHalf = lane >> 4;

  int m = mbase + ln;
  if (m >= nrows) m = nrows - 1;
  const _Float16* xr = Xh + (size_t)m * HIDF + hiHalf * 8;
  const _Float16* wf = Wp + ((size_t)(wave * 4) * 32 + lane) * 16;
  v8f c = wmma_tile(xr, wf);

  const int n    = wave * 16 + ln;
  const int mrow = mbase + hiHalf * 8;
  const float bv = bias[n];
  float* prow = P + (size_t)mrow * HIDF + n;
  float* grow = G + (size_t)mrow * HIDF + n;
  const float* dv = dinv + mrow;

  if (mbase + 16 <= nrows) {            // block-uniform fast path (no exec juggling)
#pragma unroll
    for (int v = 0; v < 8; ++v) {
      float val = c[v];
      float di  = dv[v];
      prow[(size_t)v * HIDF] = val;
      grow[(size_t)v * HIDF] = val * di * di + bv;
    }
  } else {
#pragma unroll
    for (int v = 0; v < 8; ++v) {
      int r = mrow + v;
      if (r < nrows) {
        float val = c[v];
        float di  = dv[v];
        prow[(size_t)v * HIDF] = val;
        grow[(size_t)v * HIDF] = val * di * di + bv;
      }
    }
  }
}

// Final GEMM: Y[nrows x 64] = Xh @ Wf + bf   (4 waves per block)
__global__ void gemm_final_k(const _Float16* __restrict__ Xh,
                             const _Float16* __restrict__ Wp,
                             const float* __restrict__ bias,
                             float* __restrict__ Y, int nrows) {
  constexpr int NOUT = 64;
  const int wave   = threadIdx.x >> 5;
  const int lane   = threadIdx.x & 31;
  const int mbase  = blockIdx.x * 16;
  const int ln     = lane & 15;
  const int hiHalf = lane >> 4;

  int m = mbase + ln;
  if (m >= nrows) m = nrows - 1;
  const _Float16* xr = Xh + (size_t)m * HIDF + hiHalf * 8;
  const _Float16* wf = Wp + ((size_t)(wave * 4) * 32 + lane) * 16;
  v8f c = wmma_tile(xr, wf);

  const int n    = wave * 16 + ln;
  const int mrow = mbase + hiHalf * 8;
  const float bv = bias[n];
  float* yrow = Y + (size_t)mrow * NOUT + n;

  if (mbase + 16 <= nrows) {
#pragma unroll
    for (int v = 0; v < 8; ++v) yrow[(size_t)v * NOUT] = c[v] + bv;
  } else {
#pragma unroll
    for (int v = 0; v < 8; ++v) {
      int r = mrow + v;
      if (r < nrows) yrow[(size_t)v * NOUT] = c[v] + bv;
    }
  }
}

// ---------------------------------------------------------------------------
// Degree / dinv
// ---------------------------------------------------------------------------
__global__ void zero_k(float* __restrict__ p, int n) {
  int i = blockIdx.x * blockDim.x + threadIdx.x;
  if (i < n) p[i] = 0.0f;
}

__global__ void deg_k(const long long* __restrict__ ei, float* __restrict__ deg, int E) {
  int e = blockIdx.x * blockDim.x + threadIdx.x;
  if (e < E) unsafeAtomicAdd(&deg[(int)ei[(size_t)E + e]], 1.0f);
}

__global__ void dinv_k(float* __restrict__ d, int n) {
  int i = blockIdx.x * blockDim.x + threadIdx.x;
  if (i < n) d[i] = rsqrtf(d[i] + 1.0f);
}

// Edge scatter: one wave per edge; 32 lanes x float4 = 128-float row.
__global__ void edge_k(const long long* __restrict__ ei, const float* __restrict__ H,
                       const float* __restrict__ dinv, float* __restrict__ G, int E) {
  int e    = blockIdx.x * (blockDim.x >> 5) + (threadIdx.x >> 5);
  int lane = threadIdx.x & 31;
  if (e >= E) return;
  int src = (int)ei[e];
  int dst = (int)ei[(size_t)E + e];
  float nm = dinv[src] * dinv[dst];
  const float4 h = *(const float4*)(H + (size_t)src * HIDF + lane * 4);
  float* g = G + (size_t)dst * HIDF + lane * 4;
  unsafeAtomicAdd(g + 0, h.x * nm);
  unsafeAtomicAdd(g + 1, h.y * nm);
  unsafeAtomicAdd(g + 2, h.z * nm);
  unsafeAtomicAdd(g + 3, h.w * nm);
}

// ReLU with selectable fp32 (attention inputs) and/or f16 (next GEMM) outputs
template <bool F32OUT, bool F16OUT>
__global__ void relu_k(const float* __restrict__ G, float* __restrict__ o32,
                       _Float16* __restrict__ o16, int n32) {
  int t = blockIdx.x * blockDim.x + threadIdx.x;
  if (t >= n32) return;
  float4 g = *(const float4*)(G + ((size_t)t << 2));
  g.x = fmaxf(g.x, 0.0f); g.y = fmaxf(g.y, 0.0f);
  g.z = fmaxf(g.z, 0.0f); g.w = fmaxf(g.w, 0.0f);
  if (F32OUT) *(float4*)(o32 + ((size_t)t << 2)) = g;
  if (F16OUT) {
    v4h h = { (_Float16)g.x, (_Float16)g.y, (_Float16)g.z, (_Float16)g.w };
    *(v4h*)(o16 + ((size_t)t << 2)) = h;
  }
}

// ---------------------------------------------------------------------------
// Attention combine: one wave per node; emits f16 row-major (final GEMM input)
// ---------------------------------------------------------------------------
__device__ __forceinline__ float dot4(float4 a, float4 b) {
  return a.x * b.x + a.y * b.y + a.z * b.z + a.w * b.w;
}

__global__ void attn_k(const float* __restrict__ h1, const float* __restrict__ h2,
                       const float* __restrict__ h3, const float* __restrict__ Wa,
                       const float* __restrict__ ba, _Float16* __restrict__ Hout,
                       int n) {
  int node = blockIdx.x * (blockDim.x >> 5) + (threadIdx.x >> 5);
  int lane = threadIdx.x & 31;
  if (node >= n) return;
  size_t off = (size_t)node * HIDF + lane * 4;
  float4 a1 = *(const float4*)(h1 + off);
  float4 a2 = *(const float4*)(h2 + off);
  float4 a3 = *(const float4*)(h3 + off);
  float4 wv = *(const float4*)(Wa + lane * 4);
  float s1 = dot4(a1, wv), s2 = dot4(a2, wv), s3 = dot4(a3, wv);
#pragma unroll
  for (int o = 16; o > 0; o >>= 1) {
    s1 += __shfl_xor(s1, o, 32);
    s2 += __shfl_xor(s2, o, 32);
    s3 += __shfl_xor(s3, o, 32);
  }
  float b0 = ba[0];
  s1 += b0; s2 += b0; s3 += b0;
  float mx = fmaxf(s1, fmaxf(s2, s3));
  float e1 = __expf(s1 - mx), e2 = __expf(s2 - mx), e3 = __expf(s3 - mx);
  float inv = 1.0f / (e1 + e2 + e3);
  float w1 = e1 * inv, w2 = e2 * inv, w3 = e3 * inv;
  v4h o4 = { (_Float16)(w1 * a1.x + w2 * a2.x + w3 * a3.x + a1.x),
             (_Float16)(w1 * a1.y + w2 * a2.y + w3 * a3.y + a1.y),
             (_Float16)(w1 * a1.z + w2 * a2.z + w3 * a3.z + a1.z),
             (_Float16)(w1 * a1.w + w2 * a2.w + w3 * a3.w + a1.w) };
  *(v4h*)(Hout + off) = o4;
}

// ---------------------------------------------------------------------------
extern "C" void kernel_launch(void* const* d_in, const int* in_sizes, int n_in,
                              void* d_out, int out_size, void* d_ws, size_t ws_size,
                              hipStream_t stream) {
  const float*     x  = (const float*)d_in[0];
  const long long* ei = (const long long*)d_in[1];
  const float* W[6]  = { (const float*)d_in[2],  (const float*)d_in[4],
                         (const float*)d_in[6],  (const float*)d_in[8],
                         (const float*)d_in[10], (const float*)d_in[12] };
  const float* B[6]  = { (const float*)d_in[3],  (const float*)d_in[5],
                         (const float*)d_in[7],  (const float*)d_in[9],
                         (const float*)d_in[11], (const float*)d_in[13] };
  const float *Wa = (const float*)d_in[14], *ba = (const float*)d_in[15];
  const float *Wf = (const float*)d_in[16], *bf = (const float*)d_in[17];

  const int N   = in_sizes[0] / HIDF;
  const int E   = in_sizes[1] / 2;
  const int n32 = N * 32;

  // ---- workspace layout (fp32 region, then f16 region) ----
  float* ws = (float*)d_ws;
  const size_t NF     = (size_t)N * HIDF;
  const size_t nAlign = ((size_t)N + 3) & ~(size_t)3;
  float* dinv = ws;
  float* P  = ws + nAlign;
  float* G  = P  + NF;
  float* h1 = G  + NF;
  float* h2 = h1 + NF;
  float* h3 = h2 + NF;
  _Float16* xh  = (_Float16*)(h3 + NF);   // f16 copy of x
  _Float16* t16 = xh + NF;                // chain activations; reused by attn
  _Float16* wp  = t16 + NF;               // packed weights
  _Float16* wpk[6];
  for (int i = 0; i < 6; ++i) wpk[i] = wp + (size_t)i * 128 * 128;
  _Float16* wpf = wp + (size_t)6 * 128 * 128;

  const int T = 256;
  const int gemmGrid = (N + 15) / 16;

  // ---- per-launch prep: pack weights, convert x, degree->dinv ----
  for (int i = 0; i < 6; ++i)
    pack_w_k<<<4, 256, 0, stream>>>(W[i], wpk[i], 128);
  pack_w_k<<<2, 256, 0, stream>>>(Wf, wpf, 64);
  cvt16_k<<<(n32 + T - 1) / T, T, 0, stream>>>(x, xh, n32);
  zero_k<<<(N + T - 1) / T, T, 0, stream>>>(dinv, N);
  deg_k <<<(E + T - 1) / T, T, 0, stream>>>(ei, dinv, E);
  dinv_k<<<(N + T - 1) / T, T, 0, stream>>>(dinv, N);

  // ---- GCN layer: fused GEMM(+self-loop init), edge scatter, ReLU ----
  auto gcn = [&](const _Float16* xin, int wi, float* out32, _Float16* out16) {
    gemm_gcn_k<<<gemmGrid, 256, 0, stream>>>(xin, wpk[wi], B[wi], dinv, P, G, N);
    edge_k<<<(E + 7) / 8, T, 0, stream>>>(ei, P, dinv, G, E);
    int g = (n32 + T - 1) / T;
    if (out32 && out16)
      relu_k<true,  true ><<<g, T, 0, stream>>>(G, out32, out16, n32);
    else if (out32)
      relu_k<true,  false><<<g, T, 0, stream>>>(G, out32, out16, n32);
    else
      relu_k<false, true ><<<g, T, 0, stream>>>(G, out32, out16, n32);
  };

  gcn(xh,  0, h1,      nullptr);   // h1 (fp32, attention + residual)
  gcn(xh,  1, nullptr, t16);       // t  (f16 only, feeds next GEMM)
  gcn(t16, 2, h2,      nullptr);   // h2 (fp32, attention)
  gcn(xh,  3, nullptr, t16);       // t
  gcn(t16, 4, nullptr, t16);       // t (GEMM consumes t16 before ReLU rewrites)
  gcn(t16, 5, h3,      nullptr);   // h3 (fp32, attention)

  // attention combine -> t16 (f16), then final 128->64 GEMM -> d_out
  attn_k<<<(N + 7) / 8, T, 0, stream>>>(h1, h2, h3, Wa, ba, t16, N);
  gemm_final_k<<<gemmGrid, 128, 0, stream>>>(t16, wpf, bf, (float*)d_out, N);
}
